// HSPMNBlock_59064390254755
// MI455X (gfx1250) — matrix operand order repaired and checked by reading, not compiled, not run on hardware
//
#include <hip/hip_runtime.h>
#include <math.h>

#define BB   2
#define SS   2048
#define DD   1024
#define HH   16
#define HKVN 4
#define HDIM 64
#define MLPD 4096
#define KTOK 409
#define MTOT (BB * SS) // 4096

typedef __attribute__((ext_vector_type(16))) __bf16        v16bf;
typedef __attribute__((ext_vector_type(8)))  float         v8f;
typedef __attribute__((ext_vector_type(4)))  unsigned int  v4u;

union FragBF { v16bf v; v4u q[2]; unsigned short u[16]; };

__device__ inline unsigned short f2bf(float f) {
  unsigned int x = __float_as_uint(f);
  unsigned int r = (x + 0x7FFFu + ((x >> 16) & 1u)) >> 16;
  return (unsigned short)r;
}
__device__ inline float bf2f(unsigned short h) {
  return __uint_as_float(((unsigned int)h) << 16);
}

// ============================= small kernels =============================

// transpose + cast: src [K][N] f32 -> dst [N][K] bf16
__global__ void cast_t_f32_bf16(const float* __restrict__ src,
                                unsigned short* __restrict__ dst, int K, int N) {
  size_t i = (size_t)blockIdx.x * blockDim.x + threadIdx.x;
  if (i >= (size_t)K * N) return;
  int k = (int)(i % K);
  size_t n = i / K;
  dst[i] = f2bf(src[(size_t)k * N + n]);
}

// v [b][s][kv*64+d] bf16 -> vT [b][kv][d][s] bf16
__global__ void vtrans_kernel(const unsigned short* __restrict__ v,
                              unsigned short* __restrict__ vT) {
  size_t i = (size_t)blockIdx.x * blockDim.x + threadIdx.x;
  if (i >= (size_t)MTOT * (HKVN * HDIM)) return;
  int s = (int)(i % SS);
  size_t r = i / SS;
  int d = (int)(r % HDIM);
  size_t r2 = r / HDIM;
  int kv = (int)(r2 % HKVN);
  int b = (int)(r2 / HKVN);
  vT[i] = v[((size_t)(b * SS + s)) * (HKVN * HDIM) + kv * HDIM + d];
}

__global__ __launch_bounds__(256)
void router_kernel(const float* __restrict__ x, const float* __restrict__ gw,
                   const float* __restrict__ gb, int* __restrict__ sel,
                   int* __restrict__ selCnt, float* __restrict__ accum) {
  __shared__ float ls[SS];
  __shared__ float red[256];
  __shared__ int   ired[256];
  int b = blockIdx.x, tid = threadIdx.x;
  for (int s = tid; s < SS; s += 256) {
    const float* xr = x + ((size_t)b * SS + s) * DD;
    float a = 0.f;
    for (int d = 0; d < DD; ++d) a += xr[d] * gw[d];
    ls[s] = a + gb[0];
  }
  __syncthreads();
  float sp = 0.f, se = 0.f;
  for (int s = tid; s < SS; s += 256) {
    float p = 1.f / (1.f + expf(-ls[s]));
    sp += p;
    se += -(p * logf(p + 1e-10f) + (1.f - p) * logf(1.f - p + 1e-10f));
  }
  red[tid] = sp; __syncthreads();
  for (int o = 128; o > 0; o >>= 1) { if (tid < o) red[tid] += red[tid + o]; __syncthreads(); }
  if (tid == 0) atomicAdd(&accum[0], red[0]);
  __syncthreads();
  red[tid] = se; __syncthreads();
  for (int o = 128; o > 0; o >>= 1) { if (tid < o) red[tid] += red[tid + o]; __syncthreads(); }
  if (tid == 0) atomicAdd(&accum[1], red[0]);
  __syncthreads();
  float mn = 1e30f, mx = -1e30f;
  for (int s = tid; s < SS; s += 256) { mn = fminf(mn, ls[s]); mx = fmaxf(mx, ls[s]); }
  red[tid] = mx; __syncthreads();
  for (int o = 128; o > 0; o >>= 1) { if (tid < o) red[tid] = fmaxf(red[tid], red[tid + o]); __syncthreads(); }
  float hi = red[0] + 1.f;
  __syncthreads();
  red[tid] = mn; __syncthreads();
  for (int o = 128; o > 0; o >>= 1) { if (tid < o) red[tid] = fminf(red[tid], red[tid + o]); __syncthreads(); }
  float lo = red[0] - 1.f;
  __syncthreads();
  for (int it = 0; it < 30; ++it) {
    float mid = 0.5f * (lo + hi);
    int c = 0;
    for (int s = tid; s < SS; s += 256) c += (ls[s] >= mid) ? 1 : 0;
    ired[tid] = c; __syncthreads();
    for (int o = 128; o > 0; o >>= 1) { if (tid < o) ired[tid] += ired[tid + o]; __syncthreads(); }
    int tot = ired[0];
    __syncthreads();
    if (tot >= KTOK) lo = mid; else hi = mid;
  }
  if (tid == 0) {
    float thr = lo;
    int c = 0;
    for (int s = 0; s < SS && c < KTOK; ++s)
      if (ls[s] >= thr) { sel[b * KTOK + c] = s; ++c; }
    selCnt[b] = c;
  }
}

__global__ void aux_kernel(const float* __restrict__ accum, float* __restrict__ out) {
  float mp = accum[0] / (float)(BB * SS);
  float me = accum[1] / (float)(BB * SS);
  float d = mp - 0.2f;
  out[0] = 0.1f * d * d + 0.01f * me;
}

__global__ __launch_bounds__(256)
void rmsnorm_kernel(const float* __restrict__ x, const float* __restrict__ aw,
                    const float* __restrict__ rw, unsigned short* __restrict__ xa,
                    unsigned short* __restrict__ xn) {
  int row = blockIdx.x * 8 + (threadIdx.x >> 5);
  int lane = threadIdx.x & 31;
  const float* xr = x + (size_t)row * DD;
  float ss = 0.f;
  for (int d = lane; d < DD; d += 32) { float v = xr[d]; ss += v * v; }
#pragma unroll
  for (int o = 16; o > 0; o >>= 1) ss += __shfl_xor(ss, o, 32);
  float inv = rsqrtf(ss / (float)DD + 1e-6f);
  for (int d = lane; d < DD; d += 32) {
    float v = xr[d] * inv;
    xa[(size_t)row * DD + d] = f2bf(v * aw[d]);
    xn[(size_t)row * DD + d] = f2bf(v * rw[d]);
  }
}

__global__ void conv_kernel(const unsigned short* __restrict__ xn,
                            const float* __restrict__ cw,
                            unsigned short* __restrict__ y) {
  size_t i = (size_t)blockIdx.x * blockDim.x + threadIdx.x;
  if (i >= (size_t)BB * SS * DD) return;
  int d = (int)(i % DD);
  size_t bs = i / DD;
  int s = (int)(bs % SS);
  float xm = (s > 0)      ? bf2f(xn[i - DD]) : 0.f;
  float x0 = bf2f(xn[i]);
  float xp = (s < SS - 1) ? bf2f(xn[i + DD]) : 0.f;
  y[i] = f2bf(cw[d * 3 + 0] * xm + cw[d * 3 + 1] * x0 + cw[d * 3 + 2] * xp);
}

__global__ void rope_kernel(const float* __restrict__ qF, const float* __restrict__ kF,
                            unsigned short* __restrict__ qB, unsigned short* __restrict__ kB) {
  size_t i = (size_t)blockIdx.x * blockDim.x + threadIdx.x;
  size_t tot = (size_t)BB * SS * (HH + HKVN) * 32;
  if (i >= tot) return;
  int d = (int)(i % 32);
  size_t r = i / 32;
  int hh = (int)(r % (HH + HKVN));
  size_t bs = r / (HH + HKVN);
  int s = (int)(bs % SS);
  float invf = powf(10000.f, -(float)(2 * d) / (float)HDIM);
  float ang = (float)s * invf;
  float c = cosf(ang), sn = sinf(ang);
  if (hh < HH) {
    const float* src = qF + bs * DD + (size_t)hh * HDIM;
    unsigned short* dst = qB + bs * DD + (size_t)hh * HDIM;
    float x1 = src[d], x2 = src[d + 32];
    dst[d]      = f2bf(x1 * c - x2 * sn);
    dst[d + 32] = f2bf(x2 * c + x1 * sn);
  } else {
    int kv = hh - HH;
    const float* src = kF + bs * (HKVN * HDIM) + (size_t)kv * HDIM;
    unsigned short* dst = kB + bs * (HKVN * HDIM) + (size_t)kv * HDIM;
    float x1 = src[d], x2 = src[d + 32];
    dst[d]      = f2bf(x1 * c - x2 * sn);
    dst[d + 32] = f2bf(x2 * c + x1 * sn);
  }
}

__global__ void swiglu_kernel(unsigned short* __restrict__ g,
                              const unsigned short* __restrict__ u) {
  size_t i = (size_t)blockIdx.x * blockDim.x + threadIdx.x;
  if (i >= (size_t)MTOT * MLPD) return;
  float gv = bf2f(g[i]);
  float uv = bf2f(u[i]);
  g[i] = f2bf(gv / (1.f + expf(-gv)) * uv);
}

// ============================ WMMA GEMM ============================
// C[M,N] = A[M,K] @ W[K,N]; W passed pre-transposed as WT[N][K] bf16.
// Block tile 128x128x32, double-buffered LDS (one barrier per k-step),
// 8 waves (4x2), wave tile 32x64 -> 8 WMMAs per wave per k-step.
// mode 0: bf16 out; 1: f32 out; 2: f32 + resid; 3: f32 +=
__global__ __launch_bounds__(256)
void gemm_bf16_wmma(const unsigned short* __restrict__ A,
                    const unsigned short* __restrict__ WT,
                    int M, int N, int K, int mode,
                    const float* __restrict__ resid,
                    float* __restrict__ outF,
                    unsigned short* __restrict__ outH) {
  __shared__ __align__(16) unsigned short sA[2][128][40];   // [buf][m][k]
  __shared__ __align__(16) unsigned short sBT[2][128][40];  // [buf][n][k]
  int tid = threadIdx.x;
  int lane = tid & 31;
  int w = tid >> 5;
  int wm = w >> 1, wn = w & 1;                 // 4x2 wave grid, wave tile 32x64
  int bm = blockIdx.y * 128, bn = blockIdx.x * 128;
  v8f acc[2][4];
#pragma unroll
  for (int s = 0; s < 2; ++s)
#pragma unroll
    for (int t = 0; t < 4; ++t) acc[s][t] = (v8f){};
  // staging: each tile is 128 rows x 32 cols -> 16 ushort/thread (2x b128)
  int ar = tid >> 1, ac = (tid & 1) * 16;
  int fm = lane & 15, kh = lane >> 4, kg = kh * 16;
  const unsigned short* agp = A + (size_t)(bm + ar) * K + ac;
  const unsigned short* bgp = WT + (size_t)(bn + ar) * K + ac;
  // preload k-tile 0
  {
    v4u a0 = *(const v4u*)(agp);
    v4u a1 = *(const v4u*)(agp + 8);
    v4u b0 = *(const v4u*)(bgp);
    v4u b1 = *(const v4u*)(bgp + 8);
    *(v4u*)&sA[0][ar][ac]      = a0;
    *(v4u*)&sA[0][ar][ac + 8]  = a1;
    *(v4u*)&sBT[0][ar][ac]     = b0;
    *(v4u*)&sBT[0][ar][ac + 8] = b1;
  }
  __syncthreads();
  int buf = 0;
  for (int kt = 0; kt < K; kt += 32) {
    bool more = (kt + 32) < K;
    v4u a0, a1, b0, b1;
    if (more) {  // issue next-tile global loads before compute (latency overlap)
      a0 = *(const v4u*)(agp + kt + 32);
      a1 = *(const v4u*)(agp + kt + 40);
      b0 = *(const v4u*)(bgp + kt + 32);
      b1 = *(const v4u*)(bgp + kt + 40);
      if (kt + 64 < K) {
        __builtin_prefetch((const char*)(agp + kt + 64), 0, 1);
        __builtin_prefetch((const char*)(bgp + kt + 64), 0, 1);
      }
    }
    FragBF fa[2], fb[4];
#pragma unroll
    for (int s = 0; s < 2; ++s) {
      const unsigned short* pa = &sA[buf][wm * 32 + s * 16 + fm][0];
      fa[s].q[0] = *(const v4u*)(pa + kg);
      fa[s].q[1] = *(const v4u*)(pa + 8 + kg);
    }
#pragma unroll
    for (int t = 0; t < 4; ++t) {
      const unsigned short* pb = &sBT[buf][wn * 64 + t * 16 + fm][kh * 16];
      fb[t].q[0] = *(const v4u*)pb;
      fb[t].q[1] = *(const v4u*)(pb + 8);
    }
#pragma unroll
    for (int s = 0; s < 2; ++s)
#pragma unroll
      for (int t = 0; t < 4; ++t)
        acc[s][t] = __builtin_amdgcn_wmma_f32_16x16x32_bf16(
            false, fa[s].v, false, fb[t].v, (short)0, acc[s][t], false, false);
    if (more) {
      int nb = buf ^ 1;
      *(v4u*)&sA[nb][ar][ac]      = a0;
      *(v4u*)&sA[nb][ar][ac + 8]  = a1;
      *(v4u*)&sBT[nb][ar][ac]     = b0;
      *(v4u*)&sBT[nb][ar][ac + 8] = b1;
    }
    __syncthreads();
    buf ^= 1;
  }
  int cn = lane & 15, cm = kh * 8;
#pragma unroll
  for (int s = 0; s < 2; ++s) {
    int row0 = bm + wm * 32 + s * 16 + cm;
#pragma unroll
    for (int t = 0; t < 4; ++t) {
      int col = bn + wn * 64 + t * 16 + cn;
#pragma unroll
      for (int r = 0; r < 8; ++r) {
        size_t idx = (size_t)(row0 + r) * N + col;
        float v = acc[s][t][r];
        if (mode == 0)      outH[idx] = f2bf(v);
        else if (mode == 1) outF[idx] = v;
        else if (mode == 2) outF[idx] = v + resid[idx];
        else                outF[idx] += v;
      }
    }
  }
}

// ===================== sparse flash attention (1 wave / block) =====================
// q/k fragments loaded straight from global (2x b128 each); V via pre-transposed vT.
__global__ __launch_bounds__(32)
void attn_kernel(const unsigned short* __restrict__ qB,
                 const unsigned short* __restrict__ kB,
                 const unsigned short* __restrict__ vT,
                 const int* __restrict__ sel, const int* __restrict__ selCnt,
                 unsigned short* __restrict__ ctx) {
  __shared__ __align__(16) unsigned short ps[16][40];
  __shared__ int qtok[16];
  int qt = blockIdx.x, h = blockIdx.y, b = blockIdx.z;
  int lane = threadIdx.x;
  int kvh = h >> 2;
  int cnt = selCnt[b];
  if (lane < 16) {
    int qi = qt * 16 + lane;
    qtok[lane] = (qi < cnt) ? sel[b * KTOK + qi] : -1;
  }
  __syncthreads();
  int n = lane & 15, kh = lane >> 4;
  int mh = kh * 8, kg = kh * 16;
  int maxT = -1;
#pragma unroll
  for (int i = 0; i < 16; ++i) maxT = max(maxT, qtok[i]);
  int tq8[8];
#pragma unroll
  for (int r = 0; r < 8; ++r) tq8[r] = qtok[mh + r];
  FragBF qf[2];
  {
    int tq = qtok[n];
    v4u z = {0, 0, 0, 0};
#pragma unroll
    for (int c = 0; c < 2; ++c) {
      if (tq >= 0) {
        const unsigned short* p =
            qB + ((size_t)(b * SS + tq)) * DD + h * HDIM + c * 32;
        qf[c].q[0] = *(const v4u*)(p + kg);
        qf[c].q[1] = *(const v4u*)(p + 8 + kg);
      } else {
        qf[c].q[0] = z; qf[c].q[1] = z;
      }
    }
  }
  float mrow[8], lsum[8];
#pragma unroll
  for (int r = 0; r < 8; ++r) { mrow[r] = -1e30f; lsum[r] = 0.f; }
  v8f vzero = {};
  v8f oacc[4];
#pragma unroll
  for (int g = 0; g < 4; ++g) oacc[g] = vzero;

  for (int kb = 0; kb * 32 <= maxT; ++kb) {
    int key0 = kb * 32;
    v8f s0 = vzero, s1 = vzero;
#pragma unroll
    for (int c = 0; c < 2; ++c) {
      FragBF f0, f1;
      {
        const unsigned short* p0 =
            kB + ((size_t)(b * SS + key0 + n)) * (HKVN * HDIM) + kvh * HDIM + c * 32 + kh * 16;
        f0.q[0] = *(const v4u*)p0;
        f0.q[1] = *(const v4u*)(p0 + 8);
        const unsigned short* p1 =
            kB + ((size_t)(b * SS + key0 + 16 + n)) * (HKVN * HDIM) + kvh * HDIM + c * 32 + kh * 16;
        f1.q[0] = *(const v4u*)p1;
        f1.q[1] = *(const v4u*)(p1 + 8);
      }
      s0 = __builtin_amdgcn_wmma_f32_16x16x32_bf16(false, qf[c].v, false, f0.v, (short)0, s0, false, false);
      s1 = __builtin_amdgcn_wmma_f32_16x16x32_bf16(false, qf[c].v, false, f1.v, (short)0, s1, false, false);
    }
    int keyc0 = key0 + n, keyc1 = key0 + 16 + n;
    float p0[8], p1[8];
#pragma unroll
    for (int r = 0; r < 8; ++r) {
      float a  = (tq8[r] >= 0 && keyc0 <= tq8[r]) ? s0[r] * 0.125f : -1e30f;
      float bb = (tq8[r] >= 0 && keyc1 <= tq8[r]) ? s1[r] * 0.125f : -1e30f;
      float rm = fmaxf(a, bb);
#pragma unroll
      for (int o = 1; o < 16; o <<= 1) rm = fmaxf(rm, __shfl_xor(rm, o, 32));
      float mx = fmaxf(mrow[r], rm);
      float esc = expf(mrow[r] - mx);
      float e0 = expf(a - mx), e1 = expf(bb - mx);
      float rs = e0 + e1;
#pragma unroll
      for (int o = 1; o < 16; o <<= 1) rs += __shfl_xor(rs, o, 32);
      lsum[r] = lsum[r] * esc + rs;
      mrow[r] = mx;
#pragma unroll
      for (int g = 0; g < 4; ++g) oacc[g][r] *= esc;
      p0[r] = e0; p1[r] = e1;
    }
#pragma unroll
    for (int r = 0; r < 8; ++r) {
      ps[mh + r][n]      = f2bf(p0[r]);
      ps[mh + r][16 + n] = f2bf(p1[r]);
    }
    __syncthreads();
    FragBF pa;
    {
      const unsigned short* p = &ps[n][0];
      pa.q[0] = *(const v4u*)(p + kg);
      pa.q[1] = *(const v4u*)(p + 8 + kg);
    }
#pragma unroll
    for (int g = 0; g < 4; ++g) {
      FragBF vf;
      const unsigned short* p =
          vT + (((size_t)(b * HKVN + kvh)) * HDIM + g * 16 + n) * SS + key0 + kh * 16;
      vf.q[0] = *(const v4u*)p;
      vf.q[1] = *(const v4u*)(p + 8);
      oacc[g] = __builtin_amdgcn_wmma_f32_16x16x32_bf16(false, pa.v, false, vf.v, (short)0, oacc[g], false, false);
    }
    __syncthreads();
  }
#pragma unroll
  for (int r = 0; r < 8; ++r) {
    int t = tq8[r];
    if (t < 0) continue;
    float inv = (lsum[r] > 0.f) ? 1.f / lsum[r] : 0.f;
#pragma unroll
    for (int g = 0; g < 4; ++g) {
      ctx[((size_t)(b * SS + t)) * DD + h * HDIM + g * 16 + n] = f2bf(oacc[g][r] * inv);
    }
  }
}

// ============================== launcher ==============================

extern "C" void kernel_launch(void* const* d_in, const int* in_sizes, int n_in,
                              void* d_out, int out_size, void* d_ws, size_t ws_size,
                              hipStream_t stream) {
  (void)in_sizes; (void)n_in; (void)out_size; (void)ws_size;
  const float* x       = (const float*)d_in[0];
  const float* Wq      = (const float*)d_in[1];
  const float* Wk      = (const float*)d_in[2];
  const float* Wv      = (const float*)d_in[3];
  const float* Wo      = (const float*)d_in[4];
  const float* gate_w  = (const float*)d_in[5];
  const float* gate_b  = (const float*)d_in[6];
  // d_in[7] = log_temp (monotonic; does not change top-k set)
  const float* attn_nw = (const float*)d_in[8];
  const float* ref_nw  = (const float*)d_in[9];
  const float* conv_w  = (const float*)d_in[10];
  const float* Wg      = (const float*)d_in[11];
  const float* Wu      = (const float*)d_in[12];
  const float* Wd      = (const float*)d_in[13];
  float* out = (float*)d_out;

  char* ws = (char*)d_ws;
  size_t cur = 0;
  auto alloc = [&](size_t bytes) -> void* {
    void* p = ws + cur;
    cur = (cur + bytes + 255) & ~(size_t)255;
    return p;
  };
  unsigned short* WqT  = (unsigned short*)alloc((size_t)DD * DD * 2);
  unsigned short* WkT  = (unsigned short*)alloc((size_t)DD * 256 * 2);
  unsigned short* WvT  = (unsigned short*)alloc((size_t)DD * 256 * 2);
  unsigned short* WoT  = (unsigned short*)alloc((size_t)DD * DD * 2);
  unsigned short* WgT  = (unsigned short*)alloc((size_t)DD * MLPD * 2);
  unsigned short* WuT  = (unsigned short*)alloc((size_t)DD * MLPD * 2);
  unsigned short* WdT  = (unsigned short*)alloc((size_t)MLPD * DD * 2);
  unsigned short* xaB  = (unsigned short*)alloc((size_t)MTOT * DD * 2);
  unsigned short* xnB  = (unsigned short*)alloc((size_t)MTOT * DD * 2);
  unsigned short* yB   = (unsigned short*)alloc((size_t)MTOT * DD * 2);
  float*          qF   = (float*)alloc((size_t)MTOT * DD * 4);
  float*          kF   = (float*)alloc((size_t)MTOT * 256 * 4);
  unsigned short* qBq  = (unsigned short*)alloc((size_t)MTOT * DD * 2);
  unsigned short* kBk  = (unsigned short*)alloc((size_t)MTOT * 256 * 2);
  unsigned short* vBv  = (unsigned short*)alloc((size_t)MTOT * 256 * 2);
  unsigned short* vTB  = (unsigned short*)alloc((size_t)MTOT * 256 * 2);
  unsigned short* gB   = (unsigned short*)alloc((size_t)MTOT * MLPD * 2); // reused as h
  unsigned short* uB   = (unsigned short*)alloc((size_t)MTOT * MLPD * 2);
  unsigned short* ctxB = (unsigned short*)alloc((size_t)MTOT * DD * 2);
  int*   sel    = (int*)alloc((size_t)BB * KTOK * 4);
  int*   selCnt = (int*)alloc(2 * 4);
  float* accum  = (float*)alloc(2 * 4);

  hipMemsetAsync(accum, 0, 2 * sizeof(float), stream);
  hipMemsetAsync(ctxB, 0, (size_t)MTOT * DD * 2, stream);

  auto blks = [](size_t nthreads) { return dim3((unsigned)((nthreads + 255) / 256)); };

  // weight cast + transpose (f32 [K][N] -> bf16 [N][K])
  cast_t_f32_bf16<<<blks((size_t)DD * DD),   256, 0, stream>>>(Wq, WqT, DD, DD);
  cast_t_f32_bf16<<<blks((size_t)DD * 256),  256, 0, stream>>>(Wk, WkT, DD, 256);
  cast_t_f32_bf16<<<blks((size_t)DD * 256),  256, 0, stream>>>(Wv, WvT, DD, 256);
  cast_t_f32_bf16<<<blks((size_t)DD * DD),   256, 0, stream>>>(Wo, WoT, DD, DD);
  cast_t_f32_bf16<<<blks((size_t)DD * MLPD), 256, 0, stream>>>(Wg, WgT, DD, MLPD);
  cast_t_f32_bf16<<<blks((size_t)DD * MLPD), 256, 0, stream>>>(Wu, WuT, DD, MLPD);
  cast_t_f32_bf16<<<blks((size_t)MLPD * DD), 256, 0, stream>>>(Wd, WdT, MLPD, DD);

  router_kernel<<<BB, 256, 0, stream>>>(x, gate_w, gate_b, sel, selCnt, accum);
  rmsnorm_kernel<<<MTOT / 8, 256, 0, stream>>>(x, attn_nw, ref_nw, xaB, xnB);
  conv_kernel<<<blks((size_t)MTOT * DD), 256, 0, stream>>>(xnB, conv_w, yB);

  // projections (M = 4096 = 32 * 128; all N multiples of 128)
  gemm_bf16_wmma<<<dim3(DD / 128, MTOT / 128), 256, 0, stream>>>(
      xaB, WqT, MTOT, DD, DD, 1, nullptr, qF, nullptr);
  gemm_bf16_wmma<<<dim3(256 / 128, MTOT / 128), 256, 0, stream>>>(
      xaB, WkT, MTOT, 256, DD, 1, nullptr, kF, nullptr);
  gemm_bf16_wmma<<<dim3(256 / 128, MTOT / 128), 256, 0, stream>>>(
      xaB, WvT, MTOT, 256, DD, 0, nullptr, nullptr, vBv);
  gemm_bf16_wmma<<<dim3(MLPD / 128, MTOT / 128), 256, 0, stream>>>(
      yB, WgT, MTOT, MLPD, DD, 0, nullptr, nullptr, gB);
  gemm_bf16_wmma<<<dim3(MLPD / 128, MTOT / 128), 256, 0, stream>>>(
      yB, WuT, MTOT, MLPD, DD, 0, nullptr, nullptr, uB);

  rope_kernel<<<blks((size_t)BB * SS * (HH + HKVN) * 32), 256, 0, stream>>>(qF, kF, qBq, kBk);
  vtrans_kernel<<<blks((size_t)MTOT * 256), 256, 0, stream>>>(vBv, vTB);
  swiglu_kernel<<<blks((size_t)MTOT * MLPD), 256, 0, stream>>>(gB, uB);

  attn_kernel<<<dim3((KTOK + 15) / 16, HH, BB), 32, 0, stream>>>(
      qBq, kBk, vTB, sel, selCnt, ctxB);

  // out = x + MLP(h)   then   out += ctx @ Wo
  gemm_bf16_wmma<<<dim3(DD / 128, MTOT / 128), 256, 0, stream>>>(
      gB, WdT, MTOT, DD, MLPD, 2, x, out, nullptr);
  gemm_bf16_wmma<<<dim3(DD / 128, MTOT / 128), 256, 0, stream>>>(
      ctxB, WoT, MTOT, DD, DD, 3, nullptr, out, nullptr);

  aux_kernel<<<1, 1, 0, stream>>>(accum, out + (size_t)MTOT * DD);
}
